// LightCurveRNN_1657857376385
// MI455X (gfx1250) — matrix operand order repaired and checked by reading, not compiled, not run on hardware
//
#include <hip/hip_runtime.h>

typedef __attribute__((ext_vector_type(16))) _Float16 v16h;
typedef __attribute__((ext_vector_type(8)))  _Float16 v8h;
typedef __attribute__((ext_vector_type(8)))  float    v8f;

#define B_SZ 1024
#define T_SZ 1024
#define H_SZ 128
#define C_SZ 14

// CDNA5 native transcendental tanh (v_tanh_f32); branch-free fallback.
#if __has_builtin(__builtin_amdgcn_tanhf)
__device__ __forceinline__ float fast_tanh(float x) { return __builtin_amdgcn_tanhf(x); }
#else
__device__ __forceinline__ float fast_tanh(float x) {
    float e = __builtin_amdgcn_exp2f(x * 2.885390081777927f);
    return 1.0f - 2.0f * __builtin_amdgcn_rcpf(e + 1.0f);
}
#endif

__device__ __forceinline__ v16h cat8(v8h lo, v8h hi) {
    return __builtin_shufflevector(lo, hi, 0,1,2,3,4,5,6,7,8,9,10,11,12,13,14,15);
}

// A fragment (16x32 f16, M x K): lane<16 holds row l16, K = kt*32 + 0..15;
// lane>=16 holds row l16, K = kt*32 + 16..31. hbuf row-major [16][128] f16.
__device__ __forceinline__ v16h load_a(const _Float16* hbuf, int kt, int l16, int half_id) {
    const _Float16* p = hbuf + l16 * H_SZ + kt * 32 + half_id * 16;
    v8h lo = *(const v8h*)(p);
    v8h hi = *(const v8h*)(p + 8);
    return cat8(lo, hi);
}

// B fragment (32x16 f16, K x N) for h @ W^T from f16 LDS copy of W (row-major).
// Lane holds column n; lanes 0-15 carry K = kt*32+0..15, lanes 16-31 K = kt*32+16..31.
__device__ __forceinline__ v16h load_b(const _Float16* w, int nt, int kt, int l16, int half_id) {
    const _Float16* p = w + (nt * 16 + l16) * H_SZ + kt * 32 + half_id * 16;
    v8h lo = *(const v8h*)(p);
    v8h hi = *(const v8h*)(p + 8);
    return cat8(lo, hi);
}

// B fragment built straight from the fp32 global weight matrix (init-time only).
__device__ __forceinline__ v16h make_b_global(const float* w, int nt, int kt, int l16, int half_id) {
    const float* p = w + (size_t)(nt * 16 + l16) * H_SZ + kt * 32 + half_id * 16;
    v16h r;
    #pragma unroll
    for (int i = 0; i < 16; ++i) r[i] = (_Float16)p[i];
    return r;
}

__global__ __launch_bounds__(128)
void rnn2_fused_wmma(const float* __restrict__ x,
                     const float* __restrict__ Wih0,
                     const float* __restrict__ Whh0,
                     const float* __restrict__ bih0,
                     const float* __restrict__ bhh0,
                     const float* __restrict__ Wih1,
                     const float* __restrict__ Whh1,
                     const float* __restrict__ bih1,
                     const float* __restrict__ bhh1,
                     const float* __restrict__ fcw,
                     const float* __restrict__ fcb,
                     float* __restrict__ out)
{
    __shared__ __align__(16) _Float16 sWih1[H_SZ * H_SZ];   // 32 KB, loads overlap section-B A loads
    __shared__ __align__(16) _Float16 sH1[2][16 * H_SZ];    // ping-pong layer-0 hidden (8 KB)
    __shared__ __align__(16) _Float16 sH2[2][16 * H_SZ];    // ping-pong layer-1 hidden (8 KB)

    const int tid     = threadIdx.x;
    const int wave    = tid >> 5;          // 0..3: owns N-tiles {2w, 2w+1}
    const int lane    = tid & 31;
    const int l16     = lane & 15;
    const int half_id = (lane >> 4) & 1;
    const int b0      = blockIdx.x * 16;   // this block's 16 batch rows

    // ---- stage Wih1 fp32 -> f16 into LDS; zero h buffers
    for (int i = tid; i < H_SZ * H_SZ; i += 128) sWih1[i] = (_Float16)Wih1[i];
    for (int i = tid; i < 16 * H_SZ; i += 128) {
        sH1[0][i] = (_Float16)0.f; sH1[1][i] = (_Float16)0.f;
        sH2[0][i] = (_Float16)0.f; sH2[1][i] = (_Float16)0.f;
    }

    // ---- recurrence weights register-resident: this wave's 2 N-tiles x 4 K-tiles
    v16h Bhh0[2][4], Bhh1[2][4];
    #pragma unroll
    for (int j = 0; j < 2; ++j) {
        const int nt = wave * 2 + j;
        #pragma unroll
        for (int kt = 0; kt < 4; ++kt) {
            Bhh0[j][kt] = make_b_global(Whh0, nt, kt, l16, half_id);
            Bhh1[j][kt] = make_b_global(Whh1, nt, kt, l16, half_id);
        }
    }

    // ---- x/bias fold operands (layer-0 input projection as one extra WMMA K-tile):
    // B5 rows: k=0 -> W_ih0[n], k=1 -> b_ih0[n]+b_hh0[n], rest 0 (and all-zero for lanes>=16,
    // which also nullifies the garbage K=16..31 lanes of A5).
    v16h B5[2];
    float b1n[2];
    #pragma unroll
    for (int j = 0; j < 2; ++j) {
        const int n = (wave * 2 + j) * 16 + l16;
        v16h b = {};
        if (half_id == 0) {
            b[0] = (_Float16)Wih0[n];
            b[1] = (_Float16)(bih0[n] + bhh0[n]);
        }
        B5[j] = b;
        b1n[j] = bih1[n] + bhh1[n];
    }
    v16h A5 = {};
    A5[1] = (_Float16)1.0f;   // bias row; A5[0] set per step to x
    __syncthreads();

    v16h Ah1[4], Ah2[4];

    const float* xrow = x + (size_t)(b0 + l16) * T_SZ;
    float xv = xrow[0];                       // software-pipelined x value for step t

    #pragma unroll 2
    for (int t = 0; t < T_SZ; ++t) {
        const int p = t & 1;                  // read parity; write to 1-p (static under unroll-2)

        // ======== section A (pre-barrier) ========
        A5[0] = (_Float16)xv;                 // lanes>=16 masked by B5=0

        // prefetch next step's x (latency hidden behind this whole step)
        const int tn = (t + 1 < T_SZ) ? (t + 1) : (T_SZ - 1);
        const float xv_next = xrow[tn];

        #pragma unroll
        for (int kt = 0; kt < 4; ++kt) {
            Ah1[kt] = load_a(sH1[p], kt, l16, half_id);  // h1_{t-1}
            Ah2[kt] = load_a(sH2[p], kt, l16, half_id);  // h2_{t-1}
        }

        // layer 0: h1_t = tanh([x,1] @ B5 + h1 @ Whh0^T)   (all-register B operands)
        #pragma unroll
        for (int j = 0; j < 2; ++j) {
            const int nt = wave * 2 + j;
            v8f acc = {};
            acc = __builtin_amdgcn_wmma_f32_16x16x32_f16(
                      false, A5, false, B5[j], (short)0, acc, false, false);
            #pragma unroll
            for (int kt = 0; kt < 4; ++kt)
                acc = __builtin_amdgcn_wmma_f32_16x16x32_f16(
                          false, Ah1[kt], false, Bhh0[j][kt], (short)0, acc, false, false);
            #pragma unroll
            for (int v = 0; v < 8; ++v) {
                float hv = fast_tanh(acc[v]);
                sH1[1 - p][(v + 8 * half_id) * H_SZ + nt * 16 + l16] = (_Float16)hv;
            }
        }

        // layer 1 partial: acc1 = b1 + h2_{t-1} @ Whh1^T  (independent of h1_t -> pre-barrier)
        v8f acc1[2];
        #pragma unroll
        for (int j = 0; j < 2; ++j) {
            #pragma unroll
            for (int v = 0; v < 8; ++v) acc1[j][v] = b1n[j];
            #pragma unroll
            for (int kt = 0; kt < 4; ++kt)
                acc1[j] = __builtin_amdgcn_wmma_f32_16x16x32_f16(
                              false, Ah2[kt], false, Bhh1[j][kt], (short)0, acc1[j], false, false);
        }
        __syncthreads();   // h1_t visible to all waves

        // ======== section B (post-barrier): only the Wih1 half of layer 1 ========
        #pragma unroll
        for (int kt = 0; kt < 4; ++kt) Ah1[kt] = load_a(sH1[1 - p], kt, l16, half_id); // h1_t
        #pragma unroll
        for (int j = 0; j < 2; ++j) {
            const int nt = wave * 2 + j;
            v8f acc = acc1[j];
            #pragma unroll
            for (int kt = 0; kt < 4; ++kt) {
                v16h b = load_b(sWih1, nt, kt, l16, half_id);
                acc = __builtin_amdgcn_wmma_f32_16x16x32_f16(
                          false, Ah1[kt], false, b, (short)0, acc, false, false);
            }
            #pragma unroll
            for (int v = 0; v < 8; ++v) {
                float hv = fast_tanh(acc[v]);
                sH2[1 - p][(v + 8 * half_id) * H_SZ + nt * 16 + l16] = (_Float16)hv;
            }
        }
        __syncthreads();   // h2_t visible; fences next step's loads vs stores

        xv = xv_next;      // rotate software pipeline
    }

    // ---- FC epilogue: final h2 is in the buffer written at t = T-1
    const _Float16* hfin = sH2[1 - ((T_SZ - 1) & 1)];
    for (int idx = tid; idx < 16 * C_SZ; idx += 128) {
        int m = idx / C_SZ, c = idx - m * C_SZ;
        float s = fcb[c];
        #pragma unroll 4
        for (int k = 0; k < H_SZ; ++k) s += (float)hfin[m * H_SZ + k] * fcw[c * H_SZ + k];
        out[(size_t)(b0 + m) * C_SZ + c] = s;
    }
}

extern "C" void kernel_launch(void* const* d_in, const int* in_sizes, int n_in,
                              void* d_out, int out_size, void* d_ws, size_t ws_size,
                              hipStream_t stream) {
    (void)in_sizes; (void)n_in; (void)d_ws; (void)ws_size; (void)out_size;
    const float* x    = (const float*)d_in[0];
    const float* Wih0 = (const float*)d_in[1];
    const float* Whh0 = (const float*)d_in[2];
    const float* bih0 = (const float*)d_in[3];
    const float* bhh0 = (const float*)d_in[4];
    const float* Wih1 = (const float*)d_in[5];
    const float* Whh1 = (const float*)d_in[6];
    const float* bih1 = (const float*)d_in[7];
    const float* bhh1 = (const float*)d_in[8];
    const float* fcw  = (const float*)d_in[9];
    const float* fcb  = (const float*)d_in[10];
    float* out = (float*)d_out;

    dim3 grid(B_SZ / 16);   // 64 blocks, each owns 16 batch rows
    dim3 block(128);        // 4 waves cooperate on the 8 column tiles
    hipLaunchKernelGGL(rnn2_fused_wmma, grid, block, 0, stream,
                       x, Wih0, Whh0, bih0, bhh0, Wih1, Whh1, bih1, bhh1, fcw, fcb, out);
}
